// MultiHeadAttention_24017457119732
// MI455X (gfx1250) — matrix operand order repaired and checked by reading, not compiled, not run on hardware
//
#include <hip/hip_runtime.h>

#define D_MODEL   1024
#define NUM_HEADS 16
#define HEAD_DIM  64
#define BATCH     4
#define SEQ       2048
#define ROWS      (BATCH * SEQ)   // 8192

typedef __attribute__((ext_vector_type(16))) __bf16 v16bf;
typedef __attribute__((ext_vector_type(8)))  float  v8f;

union Frag { v16bf h; unsigned int u[8]; };

// ---------- helpers ----------

__device__ __forceinline__ unsigned short f2bf(float f) {
  unsigned int b = __float_as_uint(f);
  return (unsigned short)((b + 0x7FFFu + ((b >> 16) & 1u)) >> 16);  // RNE
}

__device__ __forceinline__ v8f wmma_bf16(v16bf a, v16bf b, v8f c) {
  // (neg_a, A, neg_b, B, c_mod, C, reuse_a, reuse_b)
  return __builtin_amdgcn_wmma_f32_16x16x32_bf16(false, a, false, b, (short)0, c,
                                                 false, false);
}

// Generic pointer to a __shared__ object: low 32 bits are the LDS byte offset
// (ISA: LDS aperture addresses truncate to addr[31:0]).
__device__ __forceinline__ unsigned lds_off(const void* p) {
  return (unsigned)(size_t)p;
}

// Async DMA of one 16-byte chunk global -> LDS (per-lane), tracked by ASYNCcnt.
__device__ __forceinline__ void async_copy16(const unsigned short* gptr,
                                             unsigned lds_addr) {
  asm volatile("global_load_async_to_lds_b128 %0, %1, off"
               :: "v"(lds_addr), "v"(gptr)
               : "memory");
}

__device__ __forceinline__ void wait_async0() {
  asm volatile("s_wait_asynccnt 0x0" ::: "memory");
}

// A fragment: 16x32 bf16 from row-major [M][K] (ushort view).
// ISA layout: lane l (row = l&15), dword j holds K = 2j + (j>=4?8:0) + (l>=16?8:0) .. +1
__device__ __forceinline__ v16bf load_frag_a(const unsigned short* p, int ld,
                                             int row0, int k0, int lane) {
  Frag f;
  const unsigned short* rp = p + (row0 + (lane & 15)) * ld + k0 + ((lane >> 4) << 3);
#pragma unroll
  for (int j = 0; j < 8; ++j) {
    int kk = 2 * j + ((j >= 4) ? 8 : 0);
    f.u[j] = *(const unsigned int*)(rp + kk);
  }
  return f.h;
}

// B fragment: 32x16 bf16; source stored row-major as [N][K] (i.e. B transposed),
// lane l holds column N = l&15, dword j holds K = k0 + 2j + (j>=4?8:0) + (l>=16?8:0) .. +1
__device__ __forceinline__ v16bf load_frag_b(const unsigned short* p, int ld,
                                             int n0, int k0, int lane) {
  Frag f;
  const unsigned short* rp = p + (n0 + (lane & 15)) * ld + k0 + ((lane >> 4) << 3);
#pragma unroll
  for (int j = 0; j < 8; ++j) f.u[j] = *(const unsigned int*)(rp + 2 * j + ((j >= 4) ? 8 : 0));
  return f.h;
}

// ---------- kernels ----------

__global__ __launch_bounds__(256) void mha_cvt_bf16(const float* __restrict__ in,
                                                    unsigned short* __restrict__ out,
                                                    int n) {
  int i = blockIdx.x * 256 + threadIdx.x;
  if (i < n) out[i] = f2bf(in[i]);
}

// C = A(bf16 [8192,1024]) x W^T(bf16 [1024,1024]) + bias
// Each wave: 32x64 output tile (2 M-tiles x 4 N-tiles), software-pipelined K loop.
// mode 0: bf16 out laid out [B,H,L,64]   (Q / K)
// mode 1: bf16 out laid out [B,H,64,L]   (V transposed, for PV B-fragments)
// mode 2: f32  out laid out [B*L, 1024]  (final projection)
__global__ __launch_bounds__(128) void mha_gemm_bf16(
    const unsigned short* __restrict__ A, const unsigned short* __restrict__ W,
    const float* __restrict__ bias, void* __restrict__ outp, int mode) {
  int lane = threadIdx.x & 31;
  int wave = threadIdx.x >> 5;
  int mg   = blockIdx.x >> 4;   // 0..63
  int ngrp = blockIdx.x & 15;   // 0..15
  int row0 = (mg * 4 + wave) * 32;  // 32 rows per wave
  int n0   = ngrp * 64;

  v8f acc[2][4] = {};

  // prologue: fragments for k0 = 0
  v16bf a0 = load_frag_a(A, D_MODEL, row0,      0, lane);
  v16bf a1 = load_frag_a(A, D_MODEL, row0 + 16, 0, lane);
  v16bf b0 = load_frag_b(W, D_MODEL, n0,      0, lane);
  v16bf b1 = load_frag_b(W, D_MODEL, n0 + 16, 0, lane);
  v16bf b2 = load_frag_b(W, D_MODEL, n0 + 32, 0, lane);
  v16bf b3 = load_frag_b(W, D_MODEL, n0 + 48, 0, lane);

  for (int k0 = 0; k0 < D_MODEL; k0 += 32) {
    // issue next K-step's loads before consuming current fragments
    int kn = (k0 + 32 < D_MODEL) ? (k0 + 32) : 0;  // last iter: harmless re-load
    v16bf na0 = load_frag_a(A, D_MODEL, row0,      kn, lane);
    v16bf na1 = load_frag_a(A, D_MODEL, row0 + 16, kn, lane);
    v16bf nb0 = load_frag_b(W, D_MODEL, n0,      kn, lane);
    v16bf nb1 = load_frag_b(W, D_MODEL, n0 + 16, kn, lane);
    v16bf nb2 = load_frag_b(W, D_MODEL, n0 + 32, kn, lane);
    v16bf nb3 = load_frag_b(W, D_MODEL, n0 + 48, kn, lane);

    acc[0][0] = wmma_bf16(a0, b0, acc[0][0]);
    acc[0][1] = wmma_bf16(a0, b1, acc[0][1]);
    acc[0][2] = wmma_bf16(a0, b2, acc[0][2]);
    acc[0][3] = wmma_bf16(a0, b3, acc[0][3]);
    acc[1][0] = wmma_bf16(a1, b0, acc[1][0]);
    acc[1][1] = wmma_bf16(a1, b1, acc[1][1]);
    acc[1][2] = wmma_bf16(a1, b2, acc[1][2]);
    acc[1][3] = wmma_bf16(a1, b3, acc[1][3]);

    a0 = na0; a1 = na1;
    b0 = nb0; b1 = nb1; b2 = nb2; b3 = nb3;
  }

  int hi = (lane >> 4) << 3;  // C/D layout: row = r + 8*(lane>=16)
  int cl = lane & 15;
#pragma unroll
  for (int m = 0; m < 2; ++m) {
#pragma unroll
    for (int t = 0; t < 4; ++t) {
      int col = n0 + t * 16 + cl;
      float bv = bias[col];
#pragma unroll
      for (int r = 0; r < 8; ++r) {
        int row = row0 + m * 16 + r + hi;
        float v = acc[m][t][r] + bv;
        if (mode == 2) {
          ((float*)outp)[(size_t)row * D_MODEL + col] = v;
        } else {
          int b = row >> 11, l = row & (SEQ - 1);
          int h = col >> 6,  d = col & (HEAD_DIM - 1);
          unsigned short* o = (unsigned short*)outp;
          if (mode == 0)
            o[((size_t)(b * NUM_HEADS + h) * SEQ + l) * HEAD_DIM + d] = f2bf(v);
          else
            o[((size_t)(b * NUM_HEADS + h) * HEAD_DIM + d) * SEQ + l] = f2bf(v);
        }
      }
    }
  }
}

// Flash attention: one block = 64 query rows of one (b,h); 4 waves x 16 rows.
// Shared K/V tiles staged global->LDS with async DMA (ASYNCcnt), double-buffered
// one key-block ahead; all 4 waves feed their WMMAs from the same LDS tile.
__global__ __launch_bounds__(128) void mha_flash_attn(
    const unsigned short* __restrict__ Q, const unsigned short* __restrict__ K,
    const unsigned short* __restrict__ VT, const int* __restrict__ mask,
    unsigned short* __restrict__ O) {
  __shared__ __align__(16) unsigned short ldsK[2][32][HEAD_DIM];  // 8 KB
  __shared__ __align__(16) unsigned short ldsV[2][HEAD_DIM][32];  // 8 KB
  __shared__ __align__(16) unsigned short ldsP[4][16][32];        // 4 KB

  int tid  = threadIdx.x;
  int lane = tid & 31;
  int wave = tid >> 5;
  const int qblocks = SEQ / 64;  // 32
  int bh   = blockIdx.x / qblocks;
  int qblk = blockIdx.x % qblocks;
  int qr0  = qblk * 64 + wave * 16;

  const unsigned short* Qb = Q  + (size_t)bh * SEQ * HEAD_DIM;
  const unsigned short* Kb = K  + (size_t)bh * SEQ * HEAD_DIM;
  const unsigned short* Vb = VT + (size_t)bh * HEAD_DIM * SEQ;
  int b = bh >> 4, h = bh & 15;

  // stage one 32-key block (K: 32x64, V: 64x32) into LDS buffer `buf`
  auto stage = [&](int buf, int kb) {
#pragma unroll
    for (int j = 0; j < 2; ++j) {
      int ch = tid + j * 128;            // 256 x 16B chunks each for K and V
      int kr = ch >> 3, kc = (ch & 7) * 8;
      async_copy16(Kb + (size_t)(kb + kr) * HEAD_DIM + kc,
                   lds_off(&ldsK[buf][kr][kc]));
      int vr = ch >> 2, vc = (ch & 3) * 8;
      async_copy16(Vb + (size_t)vr * SEQ + kb + vc,
                   lds_off(&ldsV[buf][vr][vc]));
    }
  };

  // Q tile cached in registers: 16x64 = two 16x32 A fragments
  v16bf qf0 = load_frag_a(Qb, HEAD_DIM, qr0, 0,  lane);
  v16bf qf1 = load_frag_a(Qb, HEAD_DIM, qr0, 32, lane);

  v8f acc[4] = {};
  float m_i[8], l_i[8];
#pragma unroll
  for (int r = 0; r < 8; ++r) { m_i[r] = -1e30f; l_i[r] = 0.f; }

  const float scale = 0.125f;  // 1/sqrt(64)
  int hi = (lane >> 4) << 3;
  int cl = lane & 15;

  stage(0, 0);
  wait_async0();
  __syncthreads();

  int ib = 0;
  for (int kb = 0; kb < SEQ; kb += 32, ib ^= 1) {
    if (kb + 32 < SEQ) stage(ib ^ 1, kb + 32);  // DMA next block under this math

    // S = (Q K^T) * scale, masked; two 16-key sub-tiles, fragments from LDS
    const unsigned short* Kt = &ldsK[ib][0][0];
    v8f s[2];
    {
      v8f c = {};
      c = wmma_bf16(qf0, load_frag_b(Kt, HEAD_DIM, 0, 0,  lane), c);
      c = wmma_bf16(qf1, load_frag_b(Kt, HEAD_DIM, 0, 32, lane), c);
      s[0] = c;
      c = v8f{};
      c = wmma_bf16(qf0, load_frag_b(Kt, HEAD_DIM, 16, 0,  lane), c);
      c = wmma_bf16(qf1, load_frag_b(Kt, HEAD_DIM, 16, 32, lane), c);
      s[1] = c;
    }
#pragma unroll
    for (int t = 0; t < 2; ++t) {
#pragma unroll
      for (int r = 0; r < 8; ++r) {
        int qpos = qr0 + r + hi;
        int kpos = kb + t * 16 + cl;
        float v = s[t][r] * scale;
        if (mask[(size_t)qpos * SEQ + kpos] == 0) v = -1e9f;
        s[t][r] = v;
      }
    }

    // online softmax: rows live in one 16-lane half -> xor-shuffle reductions
    float alpha[8];
#pragma unroll
    for (int r = 0; r < 8; ++r) {
      float rm = fmaxf(s[0][r], s[1][r]);
#pragma unroll
      for (int off = 1; off < 16; off <<= 1)
        rm = fmaxf(rm, __shfl_xor(rm, off, 32));
      float mn = fmaxf(m_i[r], rm);
      alpha[r] = __expf(m_i[r] - mn);
      m_i[r] = mn;
      float p0 = __expf(s[0][r] - mn);
      float p1 = __expf(s[1][r] - mn);
      ldsP[wave][r + hi][cl]      = f2bf(p0);
      ldsP[wave][r + hi][16 + cl] = f2bf(p1);
      float rs = p0 + p1;
#pragma unroll
      for (int off = 1; off < 16; off <<= 1)
        rs += __shfl_xor(rs, off, 32);
      l_i[r] = l_i[r] * alpha[r] + rs;
    }
    __syncthreads();

    // O = diag(alpha) O + P V ; P re-read from LDS in A-fragment layout,
    // V fragments from the staged LDS tile ([d][key] rows -> contiguous lane reads)
    v16bf pf = load_frag_a(&ldsP[wave][0][0], 32, 0, 0, lane);
    const unsigned short* Vt = &ldsV[ib][0][0];
    {
      v8f a;
#pragma unroll
      for (int r = 0; r < 8; ++r) a[r] = acc[0][r] * alpha[r];
      acc[0] = wmma_bf16(pf, load_frag_b(Vt, 32, 0,  0, lane), a);
#pragma unroll
      for (int r = 0; r < 8; ++r) a[r] = acc[1][r] * alpha[r];
      acc[1] = wmma_bf16(pf, load_frag_b(Vt, 32, 16, 0, lane), a);
#pragma unroll
      for (int r = 0; r < 8; ++r) a[r] = acc[2][r] * alpha[r];
      acc[2] = wmma_bf16(pf, load_frag_b(Vt, 32, 32, 0, lane), a);
#pragma unroll
      for (int r = 0; r < 8; ++r) a[r] = acc[3][r] * alpha[r];
      acc[3] = wmma_bf16(pf, load_frag_b(Vt, 32, 48, 0, lane), a);
    }

    // retire this thread's async copies, then block-wide visibility + buffer reuse
    wait_async0();
    __syncthreads();
  }

  // finalize: O /= l_i, store bf16 to [B, L, D] (head-concatenated)
#pragma unroll
  for (int r = 0; r < 8; ++r) {
    float inv = 1.f / l_i[r];
    int qpos = qr0 + r + hi;
    size_t base = ((size_t)(b * SEQ) + qpos) * D_MODEL + h * HEAD_DIM;
#pragma unroll
    for (int t = 0; t < 4; ++t)
      O[base + t * 16 + cl] = f2bf(acc[t][r] * inv);
  }
}

// ---------- launcher ----------

extern "C" void kernel_launch(void* const* d_in, const int* in_sizes, int n_in,
                              void* d_out, int out_size, void* d_ws, size_t ws_size,
                              hipStream_t stream) {
  const float* x    = (const float*)d_in[0];
  const int*   mask = (const int*)  d_in[1];
  const float* Wq   = (const float*)d_in[2];
  const float* bq   = (const float*)d_in[3];
  const float* Wk   = (const float*)d_in[4];
  const float* bk   = (const float*)d_in[5];
  const float* Wv   = (const float*)d_in[6];
  const float* bv   = (const float*)d_in[7];
  const float* Wo   = (const float*)d_in[8];
  const float* bo   = (const float*)d_in[9];

  const size_t XE = (size_t)ROWS * D_MODEL;     // 8,388,608 elems
  const size_t WE = (size_t)D_MODEL * D_MODEL;  // 1,048,576 elems

  unsigned short* xb = (unsigned short*)d_ws;   // bf16 scratch, ~88 MB total
  unsigned short* wq = xb + XE;
  unsigned short* wk = wq + WE;
  unsigned short* wv = wk + WE;
  unsigned short* wo = wv + WE;
  unsigned short* qb = wo + WE;
  unsigned short* kb = qb + XE;
  unsigned short* vt = kb + XE;
  unsigned short* ob = vt + XE;

  // 1) f32 -> bf16 conversions
  mha_cvt_bf16<<<(int)((XE + 255) / 256), 256, 0, stream>>>(x,  xb, (int)XE);
  mha_cvt_bf16<<<(int)((WE + 255) / 256), 256, 0, stream>>>(Wq, wq, (int)WE);
  mha_cvt_bf16<<<(int)((WE + 255) / 256), 256, 0, stream>>>(Wk, wk, (int)WE);
  mha_cvt_bf16<<<(int)((WE + 255) / 256), 256, 0, stream>>>(Wv, wv, (int)WE);
  mha_cvt_bf16<<<(int)((WE + 255) / 256), 256, 0, stream>>>(Wo, wo, (int)WE);

  // 2) projections (Q,K in [B,H,L,64]; V transposed to [B,H,64,L])
  mha_gemm_bf16<<<1024, 128, 0, stream>>>(xb, wq, bq, qb, 0);
  mha_gemm_bf16<<<1024, 128, 0, stream>>>(xb, wk, bk, kb, 0);
  mha_gemm_bf16<<<1024, 128, 0, stream>>>(xb, wv, bv, vt, 1);

  // 3) flash attention: B*H * (L/64) blocks
  mha_flash_attn<<<BATCH * NUM_HEADS * (SEQ / 64), 128, 0, stream>>>(qb, kb, vt,
                                                                     mask, ob);

  // 4) output projection -> f32 d_out
  mha_gemm_bf16<<<1024, 128, 0, stream>>>(ob, wo, bo, d_out, 2);
}
